// Linear_24120536334975
// MI455X (gfx1250) — matrix operand order repaired and checked by reading, not compiled, not run on hardware
//
#include <hip/hip_runtime.h>

// ---------------------------------------------------------------------------
// LoRA-fused linear for MI455X (gfx1250, wave32, WMMA).
//   out[16384,4096] = X @ W^T + 2.0 * (X @ A_a^T) @ B_a^T   (adapter a = batch)
// bf16 cast pass -> v_wmma_f32_16x16x32_bf16 GEMM with LoRA folded in as one
// extra K=32 step (rank 16 padded to 32). Main GEMM stages tiles into LDS via
// GLOBAL_LOAD_ASYNC_TO_LDS_B128 (ASYNCcnt) when the toolchain exposes it.
// Workspace: ~164 MB of d_ws (Xh/Wh/Ah/Bh/Th, see kernel_launch).
// ---------------------------------------------------------------------------

typedef __bf16  bf16x16 __attribute__((ext_vector_type(16)));
typedef float   f32x8   __attribute__((ext_vector_type(8)));
typedef float   f32x4   __attribute__((ext_vector_type(4)));
typedef unsigned int   u32x4 __attribute__((ext_vector_type(4)));
typedef unsigned short u16x4 __attribute__((ext_vector_type(4)));

union FragBF { bf16x16 bf; u32x4 q[2]; };

#define L_IN   4096
#define L_OUT  4096
#define L_M    16384   // B*S = 8*2048
#define L_RP   32      // rank 16 padded to 32

#if __has_builtin(__builtin_amdgcn_global_load_async_to_lds_b128)
#define USE_ASYNC_LDS 1
#else
#define USE_ASYNC_LDS 0
#endif

// Exact pointee type of the async-to-LDS builtins (from hipcc diagnostic):
//   param0: int __vector(4) in addrspace(1), param1: same in addrspace(3)
typedef int v4i_mem __attribute__((vector_size(16)));
typedef __attribute__((address_space(1))) v4i_mem* gbl_v4i_ptr;
typedef __attribute__((address_space(3))) v4i_mem* lds_v4i_ptr;

__device__ __forceinline__ void wait_async_zero() {
#if USE_ASYNC_LDS
#if __has_builtin(__builtin_amdgcn_s_wait_asynccnt)
  __builtin_amdgcn_s_wait_asynccnt(0);
#else
  asm volatile("s_wait_asynccnt 0" ::: "memory");
#endif
#endif
}

__device__ __forceinline__ unsigned short f2bf_rne(float f) {
  union { float f; unsigned int u; } c; c.f = f;
  unsigned int u = c.u;
  return (unsigned short)((u + 0x7FFFu + ((u >> 16) & 1u)) >> 16);
}

// ---- cast fp32 -> bf16, 4 elements/thread -------------------------------
__global__ void cast_f32_bf16_v4(const float* __restrict__ src,
                                 unsigned short* __restrict__ dst, int n4) {
  int i = blockIdx.x * blockDim.x + threadIdx.x;
  if (i >= n4) return;
  f32x4 x = ((const f32x4*)src)[i];
  u16x4 o;
  o[0] = f2bf_rne(x[0]); o[1] = f2bf_rne(x[1]);
  o[2] = f2bf_rne(x[2]); o[3] = f2bf_rne(x[3]);
  ((u16x4*)dst)[i] = o;
}

// ---- build Bh[a, n, 0..31] = bf16(2.0 * lora_b[a, n, r]) (r<16), else 0 ----
__global__ void build_bh_kernel(const float* __restrict__ lb,
                                unsigned short* __restrict__ bh) {
  int i = blockIdx.x * blockDim.x + threadIdx.x;   // over 8*4096*32
  int r   = i & 31;
  int row = i >> 5;                                 // a*OUT + n
  unsigned short v = 0;
  if (r < 16) v = f2bf_rne(2.0f * lb[row * 16 + r]);
  bh[i] = v;
}

// ---- Th[m, r] = sum_k Xh[m,k] * Ah[adapter(m), r, k]  (WMMA, rank tile) ----
__global__ void __launch_bounds__(256)
lora_t_kernel(const unsigned short* __restrict__ Xh,
              const unsigned short* __restrict__ Ah,
              unsigned short* __restrict__ Th) {
  int wid  = threadIdx.x >> 5;
  int lane = threadIdx.x & 31;
  int m0 = (blockIdx.x * 8 + wid) * 16;   // 16 rows per wave
  int adapter = m0 >> 11;                 // 2048 rows per adapter
  int lh = lane >> 4, lm = lane & 15;

  const unsigned short* xrow = Xh + (size_t)(m0 + lm) * L_IN;
  const unsigned short* arow = Ah + (size_t)adapter * 16 * L_IN + (size_t)lm * L_IN;

  f32x8 acc = {};
  for (int k = 0; k < L_IN; k += 32) {
    FragBF a, b;
    // A 16x32: lane holds row (lm), K runs [lh*8,+8) and [16+lh*8,+8)
    a.q[0] = *(const u32x4*)(xrow + k + lh * 8);
    a.q[1] = *(const u32x4*)(xrow + k + 16 + lh * 8);
    // B 32x16: lane holds column r=lm, K runs [lh*8,+8) and [16+lh*8,+8)
    b.q[0] = *(const u32x4*)(arow + k + lh * 8);
    b.q[1] = *(const u32x4*)(arow + k + 16 + lh * 8);
    acc = __builtin_amdgcn_wmma_f32_16x16x32_bf16(false, a.bf, false, b.bf,
                                                  (short)0, acc, false, false);
  }
  // C/D: lane l, vgpr v -> (M = v + 8*lh, N = lm); also zero padded cols.
#pragma unroll
  for (int v = 0; v < 8; ++v) {
    int m = m0 + v + 8 * lh;
    Th[m * L_RP + lm]      = f2bf_rne(acc[v]);
    Th[m * L_RP + 16 + lm] = 0;
  }
}

// ---- main GEMM: 128x128 tile, BK=32, double-buffered LDS, 8 waves ---------
#define BM 128
#define BN 128
#define BK 32
#define KSTEPS (L_IN / BK)   // 128 base steps + 1 LoRA step

__global__ void __launch_bounds__(256)
lora_gemm_kernel(const unsigned short* __restrict__ Xh,   // [16384,4096]
                 const unsigned short* __restrict__ Wh,   // [4096,4096]
                 const unsigned short* __restrict__ Th,   // [16384,32]
                 const unsigned short* __restrict__ Bh,   // [8,4096,32]
                 float* __restrict__ out) {               // [16384,4096]
  __shared__ alignas(16) unsigned short As[2][BM * BK];
  __shared__ alignas(16) unsigned short Bs[2][BM * BK];

  const int tid  = threadIdx.x;
  const int wid  = tid >> 5, lane = tid & 31;
  const int waveM = wid >> 1, waveN = wid & 1;          // 4 x 2 wave grid
  const int lh = lane >> 4, lm = lane & 15;
  const int mBase = blockIdx.y * BM;
  const int nBase = blockIdx.x * BN;
  const int adapter = mBase >> 11;

  // Staging: 128 rows x 4 16B-chunks = 512 chunks, 2 per thread.
  const int ar0 = tid >> 2,          ac0 = (tid & 3) * 8;
  const int ar1 = (tid + 256) >> 2,  ac1 = ((tid + 256) & 3) * 8;
  const unsigned short* bhA = Bh + (size_t)adapter * L_OUT * L_RP;

  f32x8 acc[2][4] = {};

  // -------- shared compute stage (LDS frags -> 8 WMMAs) --------------------
  auto compute = [&](int buf) {
    FragBF afr[2], bfr[4];
#pragma unroll
    for (int r = 0; r < 2; ++r) {
      int row = waveM * 32 + r * 16 + lm;
      afr[r].q[0] = *(const u32x4*)(&As[buf][row * BK + lh * 8]);
      afr[r].q[1] = *(const u32x4*)(&As[buf][row * BK + 16 + lh * 8]);
    }
#pragma unroll
    for (int c = 0; c < 4; ++c) {
      int n = waveN * 64 + c * 16 + lm;
      bfr[c].q[0] = *(const u32x4*)(&Bs[buf][n * BK + lh * 8]);
      bfr[c].q[1] = *(const u32x4*)(&Bs[buf][n * BK + 16 + lh * 8]);
    }
#pragma unroll
    for (int r = 0; r < 2; ++r)
#pragma unroll
      for (int c = 0; c < 4; ++c)
        acc[r][c] = __builtin_amdgcn_wmma_f32_16x16x32_bf16(
            false, afr[r].bf, false, bfr[c].bf, (short)0, acc[r][c],
            false, false);
  };

  // -------- source addresses for staging step s ----------------------------
  auto srcptrs = [&](int step, const unsigned short*& gA0, const unsigned short*& gA1,
                     const unsigned short*& gB0, const unsigned short*& gB1) {
    if (step < KSTEPS) {
      int k = step * BK;
      gA0 = Xh + (size_t)(mBase + ar0) * L_IN + k + ac0;
      gA1 = Xh + (size_t)(mBase + ar1) * L_IN + k + ac1;
      gB0 = Wh + (size_t)(nBase + ar0) * L_IN + k + ac0;
      gB1 = Wh + (size_t)(nBase + ar1) * L_IN + k + ac1;
    } else {  // LoRA tail: A <- Th rows, B <- Bh[adapter] rows (both K=32 wide)
      gA0 = Th + (size_t)(mBase + ar0) * L_RP + ac0;
      gA1 = Th + (size_t)(mBase + ar1) * L_RP + ac1;
      gB0 = bhA + (size_t)(nBase + ar0) * L_RP + ac0;
      gB1 = bhA + (size_t)(nBase + ar1) * L_RP + ac1;
    }
  };

#if USE_ASYNC_LDS
  // ---- async global->LDS staging (no VGPR round-trip, ASYNCcnt-tracked) ---
  auto stage = [&](int step, int buf) {
    const unsigned short *gA0, *gA1, *gB0, *gB1;
    srcptrs(step, gA0, gA1, gB0, gB1);
    __builtin_amdgcn_global_load_async_to_lds_b128(
        (gbl_v4i_ptr)gA0, (lds_v4i_ptr)&As[buf][ar0 * BK + ac0], 0, 0);
    __builtin_amdgcn_global_load_async_to_lds_b128(
        (gbl_v4i_ptr)gA1, (lds_v4i_ptr)&As[buf][ar1 * BK + ac1], 0, 0);
    __builtin_amdgcn_global_load_async_to_lds_b128(
        (gbl_v4i_ptr)gB0, (lds_v4i_ptr)&Bs[buf][ar0 * BK + ac0], 0, 0);
    __builtin_amdgcn_global_load_async_to_lds_b128(
        (gbl_v4i_ptr)gB1, (lds_v4i_ptr)&Bs[buf][ar1 * BK + ac1], 0, 0);
  };

  stage(0, 0);
  wait_async_zero();
  __syncthreads();

  for (int step = 0; step <= KSTEPS; ++step) {
    const int buf = step & 1;
    if (step < KSTEPS) stage(step + 1, buf ^ 1);  // copy overlaps WMMAs
    compute(buf);
    wait_async_zero();
    __syncthreads();
  }
#else
  // ---- fallback: global -> VGPR -> ds_write staging -----------------------
  u32x4 rA0, rA1, rB0, rB1;
  auto gload = [&](int step) {
    const unsigned short *gA0, *gA1, *gB0, *gB1;
    srcptrs(step, gA0, gA1, gB0, gB1);
    rA0 = *(const u32x4*)gA0;  rA1 = *(const u32x4*)gA1;
    rB0 = *(const u32x4*)gB0;  rB1 = *(const u32x4*)gB1;
  };
  auto sstore = [&](int buf) {
    *(u32x4*)(&As[buf][ar0 * BK + ac0]) = rA0;
    *(u32x4*)(&As[buf][ar1 * BK + ac1]) = rA1;
    *(u32x4*)(&Bs[buf][ar0 * BK + ac0]) = rB0;
    *(u32x4*)(&Bs[buf][ar1 * BK + ac1]) = rB1;
  };

  gload(0);
  sstore(0);
  __syncthreads();

  for (int step = 0; step <= KSTEPS; ++step) {
    const int buf = step & 1;
    const bool more = (step < KSTEPS);
    if (more) gload(step + 1);
    compute(buf);
    if (more) sstore(buf ^ 1);
    __syncthreads();
  }
#endif

  // Epilogue: lane l, vgpr v -> (M = v + 8*lh, N = lm)
#pragma unroll
  for (int r = 0; r < 2; ++r) {
    int mrow = mBase + waveM * 32 + r * 16 + 8 * lh;
#pragma unroll
    for (int c = 0; c < 4; ++c) {
      int ncol = nBase + waveN * 64 + c * 16 + lm;
      float* o = out + (size_t)mrow * L_OUT + ncol;
#pragma unroll
      for (int v = 0; v < 8; ++v)
        o[(size_t)v * L_OUT] = acc[r][c][v];
    }
  }
}

// ---------------------------------------------------------------------------
extern "C" void kernel_launch(void* const* d_in, const int* in_sizes, int n_in,
                              void* d_out, int out_size, void* d_ws, size_t ws_size,
                              hipStream_t stream) {
  const float* X  = (const float*)d_in[0];   // [8,2048,4096]
  const float* W  = (const float*)d_in[1];   // [4096,4096]
  const float* la = (const float*)d_in[2];   // [8,16,4096]
  const float* lb = (const float*)d_in[3];   // [8,4096,16]
  float* out = (float*)d_out;

  char* ws = (char*)d_ws;
  unsigned short* Xh = (unsigned short*)(ws);                 // 128 MiB
  unsigned short* Wh = (unsigned short*)(ws + 134217728ull);  //  32 MiB
  unsigned short* Ah = (unsigned short*)(ws + 167772160ull);  //   1 MiB
  unsigned short* Bh = (unsigned short*)(ws + 168820736ull);  //   2 MiB
  unsigned short* Th = (unsigned short*)(ws + 170917888ull);  //   1 MiB

  // 1) casts to bf16
  cast_f32_bf16_v4<<<16777216 / 256, 256, 0, stream>>>(X,  Xh, 16777216); // 67.1M elems
  cast_f32_bf16_v4<<< 4194304 / 256, 256, 0, stream>>>(W,  Wh,  4194304); // 16.8M
  cast_f32_bf16_v4<<<  131072 / 256, 256, 0, stream>>>(la, Ah,   131072); // 0.52M
  // 2) scaled + padded lora_b
  build_bh_kernel<<<1048576 / 256, 256, 0, stream>>>(lb, Bh);
  // 3) T = X @ A^T (rank tile, WMMA)
  lora_t_kernel<<<L_M / 128, 256, 0, stream>>>(Xh, Ah, Th);
  // 4) fused GEMM: out = Xh@Wh^T + Th@Bh^T (K augmented by one 32-step)
  dim3 grid(L_OUT / BN, L_M / BM);  // (32, 128)
  lora_gemm_kernel<<<grid, 256, 0, stream>>>(Xh, Wh, Th, Bh, out);
}
